// MultiheadAttention_29334626631925
// MI455X (gfx1250) — compile-verified, matrix operand order checked by reading, not run on hardware
//
#include <hip/hip_runtime.h>

typedef unsigned short u16;
typedef unsigned int   u32;

typedef __attribute__((ext_vector_type(16))) __bf16 v16bf;
typedef __attribute__((ext_vector_type(8)))  float  v8f;

#define EMB   1024
#define HEADS 16
#define HD    64
#define BATCH 4
#define SEQ   2048
#define MTOT  (BATCH*SEQ)

#define WMMA_BF16(A, B, C) \
  __builtin_amdgcn_wmma_f32_16x16x32_bf16(false, (A), false, (B), (short)0, (C), false, false)

struct U32x8 { uint4 a, b; };

__device__ __forceinline__ u16 f2bf(float f) {
  u32 u = __float_as_uint(f);
  u32 r = u + 0x7FFFu + ((u >> 16) & 1u);   // round-to-nearest-even
  return (u16)(r >> 16);
}

// Load one 16-bit WMMA operand fragment: two 16-byte chunks -> 16 bf16 per lane.
__device__ __forceinline__ v16bf ld_frag(const u16* base, int o0, int o1) {
  U32x8 t;
  t.a = *(const uint4*)(base + o0);
  t.b = *(const uint4*)(base + o1);
  return __builtin_bit_cast(v16bf, t);
}

__device__ __forceinline__ v8f v8f_splat(float x) {
  v8f r; for (int i = 0; i < 8; ++i) r[i] = x; return r;
}

__device__ __forceinline__ uint4 pack8(const float* f) {
  uint4 r;
  r.x = (u32)f2bf(f[0]) | ((u32)f2bf(f[1]) << 16);
  r.y = (u32)f2bf(f[2]) | ((u32)f2bf(f[3]) << 16);
  r.z = (u32)f2bf(f[4]) | ((u32)f2bf(f[5]) << 16);
  r.w = (u32)f2bf(f[6]) | ((u32)f2bf(f[7]) << 16);
  return r;
}

// gfx1250 async DMA: 16 bytes global -> LDS, no VGPR staging, tracked by ASYNCcnt.
__device__ __forceinline__ void async_ld_b128(u32 lds_byte_addr, const u16* gaddr) {
  asm volatile("global_load_async_to_lds_b128 %0, %1, off"
               :: "v"(lds_byte_addr), "v"(gaddr) : "memory");
}

// ---------------------------------------------------------------------------
// GEMM: out = (A @ W^T + bias) * oscale.  A:[8192,1024], W:[1024,1024] row-major.
// Software-pipelined: next tile's global loads issue before current compute.
// MODE 0: A fp32, out bf16 [B,H,S,64]      (Q / K projection; Q folds 1/sqrt(hd))
// MODE 2: A fp32, out bf16 [B,H,64,S]      (V projection, transposed)
// MODE 3: A bf16, out fp32 [M,N]           (output projection)
// ---------------------------------------------------------------------------
#define TM 128
#define TN 128
#define TK 32
#define LDSW 40   // u16 row stride: 32 data + 8 pad (bank-conflict-free b128)

template<int MODE>
__global__ __launch_bounds__(256) void gemm_bf16_wmma(const void* __restrict__ Ap,
                                                      const float* __restrict__ W,
                                                      const float* __restrict__ bias,
                                                      void* __restrict__ outp,
                                                      float oscale) {
  __shared__ u16 lA[TM * LDSW];
  __shared__ u16 lB[TN * LDSW];

  const int tid  = threadIdx.x;
  const int lane = tid & 31, wid = tid >> 5;
  const int hl = lane >> 4, lr = lane & 15;
  const int wm = wid & 3, wn = wid >> 2;           // 4x2 wave grid
  const int bm = blockIdx.x * TM, bn = blockIdx.y * TN;

  v8f acc[2][4];
  for (int nt = 0; nt < 4; ++nt) {
    float bv = bias[bn + wn*64 + nt*16 + lr];      // bias depends only on column
    acc[0][nt] = v8f_splat(bv);
    acc[1][nt] = v8f_splat(bv);
  }

  const int row = tid >> 1;            // 0..127
  const int cb  = (tid & 1) << 4;      // 0 or 16

  // per-thread staging registers
  float fa[16], fb[16];
  uint4 ua0, ua1;                      // MODE 3: bf16 A loads raw

  auto fetch_glb = [&](int k0) {
    if constexpr (MODE == 3) {
      const u16* src = (const u16*)Ap + (size_t)(bm + row) * EMB + k0 + cb;
      ua0 = *(const uint4*)(src);
      ua1 = *(const uint4*)(src + 8);
    } else {
      const float* src = (const float*)Ap + (size_t)(bm + row) * EMB + k0 + cb;
      *(float4*)&fa[0]  = *(const float4*)(src + 0);
      *(float4*)&fa[4]  = *(const float4*)(src + 4);
      *(float4*)&fa[8]  = *(const float4*)(src + 8);
      *(float4*)&fa[12] = *(const float4*)(src + 12);
    }
    const float* srcw = W + (size_t)(bn + row) * EMB + k0 + cb;
    *(float4*)&fb[0]  = *(const float4*)(srcw + 0);
    *(float4*)&fb[4]  = *(const float4*)(srcw + 4);
    *(float4*)&fb[8]  = *(const float4*)(srcw + 8);
    *(float4*)&fb[12] = *(const float4*)(srcw + 12);
  };

  fetch_glb(0);

  for (int k0 = 0; k0 < EMB; k0 += TK) {
    __syncthreads();                   // previous compute done reading LDS
    if constexpr (MODE == 3) {
      *(uint4*)&lA[row*LDSW + cb]     = ua0;
      *(uint4*)&lA[row*LDSW + cb + 8] = ua1;
    } else {
      *(uint4*)&lA[row*LDSW + cb]     = pack8(&fa[0]);
      *(uint4*)&lA[row*LDSW + cb + 8] = pack8(&fa[8]);
    }
    *(uint4*)&lB[row*LDSW + cb]     = pack8(&fb[0]);
    *(uint4*)&lB[row*LDSW + cb + 8] = pack8(&fb[8]);
    __syncthreads();

    if (k0 + TK < EMB) fetch_glb(k0 + TK);   // prefetch next tile (overlaps WMMA)

    v16bf a0 = ld_frag(&lA[(wm*32 +  0 + lr)*LDSW], hl*8, 16 + hl*8);
    v16bf a1 = ld_frag(&lA[(wm*32 + 16 + lr)*LDSW], hl*8, 16 + hl*8);
    v16bf bf[4];
    for (int nt = 0; nt < 4; ++nt)
      bf[nt] = ld_frag(&lB[(wn*64 + nt*16 + lr)*LDSW], hl*8, 16 + hl*8);
    for (int nt = 0; nt < 4; ++nt)
      acc[0][nt] = WMMA_BF16(a0, bf[nt], acc[0][nt]);
    for (int nt = 0; nt < 4; ++nt)
      acc[1][nt] = WMMA_BF16(a1, bf[nt], acc[1][nt]);
  }

  for (int mt = 0; mt < 2; ++mt)
    for (int nt = 0; nt < 4; ++nt)
      for (int v = 0; v < 8; ++v) {
        int m = bm + wm*32 + mt*16 + v + 8*hl;      // C/D layout: M = v + 8*(lane/16)
        int n = bn + wn*64 + nt*16 + lr;            //             N = lane%16
        float val = acc[mt][nt][v] * oscale;
        if constexpr (MODE == 3) {
          ((float*)outp)[(size_t)m * EMB + n] = val;
        } else {
          int bb = m >> 11, s = m & (SEQ - 1);
          int h = n >> 6, d = n & 63;
          u16 o = f2bf(val);
          if constexpr (MODE == 2)
            ((u16*)outp)[(size_t)((bb*HEADS + h)*HD + d) * SEQ + s] = o;
          else
            ((u16*)outp)[(size_t)((bb*HEADS + h)*SEQ + s) * HD + d] = o;
        }
      }
}

// ---------------------------------------------------------------------------
// Flash attention: one block per (b,h,q-tile of 64). 128 threads = 4 waves,
// each wave owns 16 q rows. Online softmax, causal; Q is pre-scaled by 1/8.
// K/V^T tiles stream global->LDS via async DMA, double-buffered (ASYNCcnt).
// q,k: [B,H,S,64] bf16.  vt: [B,H,64,S] bf16.  attn: [B,S,H*64] bf16.
// ---------------------------------------------------------------------------
#define KSTR 72   // u16 row stride for 64-wide LDS tiles (+8 pad)

__global__ __launch_bounds__(128) void flash_attn_wmma(const u16* __restrict__ q,
                                                       const u16* __restrict__ k,
                                                       const u16* __restrict__ vt,
                                                       u16* __restrict__ attn) {
  __shared__ u16 lK[2][64 * KSTR];
  __shared__ u16 lV[2][64 * KSTR];
  __shared__ u16 lP[4][16 * KSTR];

  const int qt = blockIdx.x;          // q tile (64 rows)
  const int bh = blockIdx.y;          // b*HEADS + h
  const int qbase = qt * 64;
  const u16* qh = q  + (size_t)bh * SEQ * HD;
  const u16* kh = k  + (size_t)bh * SEQ * HD;
  const u16* vh = vt + (size_t)bh * HD * SEQ;

  const int tid = threadIdx.x, lane = tid & 31, wid = tid >> 5;
  const int hl = lane >> 4, lr = lane & 15;

  // Q fragments stay in registers for the whole kernel (head_dim 64 = 2 frags)
  const u16* qr = qh + (size_t)(qbase + wid*16 + lr) * HD;
  v16bf aq0 = ld_frag(qr,      hl*8, 16 + hl*8);
  v16bf aq1 = ld_frag(qr + 32, hl*8, 16 + hl*8);

  v8f acc[4];
  float m_i[8], l_i[8];
  for (int nt = 0; nt < 4; ++nt) acc[nt] = v8f_splat(0.f);
  for (int v = 0; v < 8; ++v) { m_i[v] = -1e30f; l_i[v] = 0.f; }

  const int srow = tid >> 1;          // 0..63
  const int scb  = (tid & 1) << 5;    // 0 or 32

  // issue 8 async b128 copies: K tile row + V^T tile row -> LDS buffer `buf`
  auto stage_async = [&](int kb, int buf) {
    const u16* sk = kh + (size_t)(kb + srow) * HD + scb;
    u32 dk = (u32)(size_t)&lK[buf][srow*KSTR + scb];
    async_ld_b128(dk,      sk);
    async_ld_b128(dk + 16, sk + 8);
    async_ld_b128(dk + 32, sk + 16);
    async_ld_b128(dk + 48, sk + 24);
    const u16* sv = vh + (size_t)srow * SEQ + kb + scb;
    u32 dv = (u32)(size_t)&lV[buf][srow*KSTR + scb];
    async_ld_b128(dv,      sv);
    async_ld_b128(dv + 16, sv + 8);
    async_ld_b128(dv + 32, sv + 16);
    async_ld_b128(dv + 48, sv + 24);
  };

  const int nkt = qt + 1;             // causal: skip fully-masked key tiles
  stage_async(0, 0);

  for (int t = 0; t < nkt; ++t) {
    const int kb = t * 64;
    const int cur = t & 1;
    __syncthreads();                  // all waves done reading buffer cur^1
    if (t + 1 < nkt) {
      stage_async(kb + 64, cur ^ 1);  // DMA next tile while we compute this one
      asm volatile("s_wait_asynccnt 0x8" ::: "memory");  // tile t's 8 copies done
    } else {
      asm volatile("s_wait_asynccnt 0x0" ::: "memory");
    }
    __syncthreads();                  // tile t visible to all waves

    // preload all 8 K fragments, then issue 8 score WMMAs back-to-back
    v16bf b0[4], b1[4];
    for (int nt = 0; nt < 4; ++nt) {
      const u16* kr = &lK[cur][(nt*16 + lr)*KSTR];
      b0[nt] = ld_frag(kr,      hl*8, 16 + hl*8);
      b1[nt] = ld_frag(kr + 32, hl*8, 16 + hl*8);
    }
    v8f sc[4];
    for (int nt = 0; nt < 4; ++nt)
      sc[nt] = WMMA_BF16(aq0, b0[nt], v8f_splat(0.f));
    for (int nt = 0; nt < 4; ++nt)
      sc[nt] = WMMA_BF16(aq1, b1[nt], sc[nt]);
    // NOTE: 1/sqrt(64) already folded into Q by the projection kernel

    if (kb == qbase) {                // diagonal tile: apply causal mask
      for (int nt = 0; nt < 4; ++nt)
        for (int v = 0; v < 8; ++v) {
          int qq = wid*16 + v + 8*hl;
          int kk = nt*16 + lr;
          if (kk > qq) sc[nt][v] = -1e30f;
        }
    }

    // online softmax (row = M dim; one row's 16 cols live in one 16-lane half)
    for (int v = 0; v < 8; ++v) {
      float mv = fmaxf(fmaxf(sc[0][v], sc[1][v]), fmaxf(sc[2][v], sc[3][v]));
      for (int m = 1; m < 16; m <<= 1) mv = fmaxf(mv, __shfl_xor(mv, m, 32));
      float mn = fmaxf(m_i[v], mv);
      float scale = __expf(m_i[v] - mn);
      m_i[v] = mn;
      l_i[v] *= scale;
      for (int nt = 0; nt < 4; ++nt) acc[nt][v] *= scale;
      float rs = 0.f;
      for (int nt = 0; nt < 4; ++nt) {
        float p = __expf(sc[nt][v] - mn);
        sc[nt][v] = p;
        rs += p;
      }
      for (int m = 1; m < 16; m <<= 1) rs += __shfl_xor(rs, m, 32);
      l_i[v] += rs;
    }

    // P (C/D layout) -> LDS -> reload in A-fragment layout
    u16* pw = &lP[wid][0];
    for (int nt = 0; nt < 4; ++nt)
      for (int v = 0; v < 8; ++v)
        pw[(v + 8*hl)*KSTR + nt*16 + lr] = f2bf(sc[nt][v]);
    __syncthreads();

    // O += P @ V : preload P + all 8 V fragments, then 8 WMMAs back-to-back
    v16bf ap0 = ld_frag(&lP[wid][lr*KSTR],      hl*8, 16 + hl*8);
    v16bf ap1 = ld_frag(&lP[wid][lr*KSTR + 32], hl*8, 16 + hl*8);
    v16bf v0[4], v1[4];
    for (int nt = 0; nt < 4; ++nt) {
      const u16* vr = &lV[cur][(nt*16 + lr)*KSTR];
      v0[nt] = ld_frag(vr,      hl*8, 16 + hl*8);
      v1[nt] = ld_frag(vr + 32, hl*8, 16 + hl*8);
    }
    for (int nt = 0; nt < 4; ++nt)
      acc[nt] = WMMA_BF16(ap0, v0[nt], acc[nt]);
    for (int nt = 0; nt < 4; ++nt)
      acc[nt] = WMMA_BF16(ap1, v1[nt], acc[nt]);
  }

  // normalize and store [B,S,H*64] bf16
  const int bb = bh >> 4, h = bh & 15;
  float inv[8];
  for (int v = 0; v < 8; ++v) inv[v] = 1.0f / l_i[v];
  for (int nt = 0; nt < 4; ++nt)
    for (int v = 0; v < 8; ++v) {
      int qg = qbase + wid*16 + v + 8*hl;
      int dd = nt*16 + lr;
      attn[(size_t)(bb*SEQ + qg) * EMB + h*HD + dd] = f2bf(acc[nt][v] * inv[v]);
    }
}

// ---------------------------------------------------------------------------

extern "C" void kernel_launch(void* const* d_in, const int* in_sizes, int n_in,
                              void* d_out, int out_size, void* d_ws, size_t ws_size,
                              hipStream_t stream) {
  const float* x   = (const float*)d_in[0];
  const float* Wq  = (const float*)d_in[1];
  const float* bq  = (const float*)d_in[2];
  const float* Wk  = (const float*)d_in[3];
  const float* bk  = (const float*)d_in[4];
  const float* Wv  = (const float*)d_in[5];
  const float* bv  = (const float*)d_in[6];
  const float* Wo  = (const float*)d_in[7];
  const float* bo  = (const float*)d_in[8];
  // d_in[9] = causal_mask (mask is computed analytically in-kernel)
  float* out = (float*)d_out;
  char*  ws  = (char*)d_ws;

  const size_t seg = (size_t)BATCH * HEADS * SEQ * HD * sizeof(u16);  // 16 MB
  u16* qw = (u16*)(ws);
  u16* kw = (u16*)(ws + seg);
  u16* vw = (u16*)(ws + 2*seg);
  u16* aw = (u16*)(ws + 3*seg);

  dim3 gg(MTOT / TM, EMB / TN);   // (64, 8)
  gemm_bf16_wmma<0><<<gg, 256, 0, stream>>>(x, Wq, bq, qw, 0.125f);  // 1/sqrt(64)
  gemm_bf16_wmma<0><<<gg, 256, 0, stream>>>(x, Wk, bk, kw, 1.0f);
  gemm_bf16_wmma<2><<<gg, 256, 0, stream>>>(x, Wv, bv, vw, 1.0f);
  flash_attn_wmma<<<dim3(SEQ/64, BATCH*HEADS), 128, 0, stream>>>(qw, kw, vw, aw);
  gemm_bf16_wmma<3><<<gg, 256, 0, stream>>>(aw, Wo, bo, out, 1.0f);
}
